// PHomogeneousAutoencoder_30597347016754
// MI455X (gfx1250) — matrix operand order repaired and checked by reading, not compiled, not run on hardware
//
#include <hip/hip_runtime.h>

typedef __attribute__((ext_vector_type(16))) _Float16 v16h;
typedef __attribute__((ext_vector_type(8)))  _Float16 v8h;
typedef __attribute__((ext_vector_type(8)))  float    v8f;
typedef unsigned int v4u __attribute__((ext_vector_type(4)));
typedef unsigned int v8u __attribute__((ext_vector_type(8)));

#define HID 128
/* packed per-MLP weight block: 22528 f16 (W1t 128x32 | W2t 128x128 | W3t 16x128)
   followed by 272 f32 biases (b1 128 | b2 128 | b3 16) = 46144 bytes = 5768 x 8B */
#define WB_HALFS     22528
#define WB_BIAS_OFF  45056
#define WB_BYTES     46144
#define WB_QWORDS    5768
#define SCAL_COLS    20

/* shared memory carve-up (bytes): double-buffered weights + ping-pong activations */
#define SM_BUF0_OFF  0
#define SM_BUF1_OFF  46144
#define SM_H1_OFF    92288
#define SM_H2_OFF    125056
#define SM_SCAL_OFF  157824
#define SMEM_BYTES   168064

/* file-scope dynamic LDS: referencing it directly inside device functions keeps
   every derived pointer in addrspace(3) (pure 32-bit ds addressing, no generic
   pointer null-check cndmasks across the noinline call boundary) */
extern __shared__ char smem[];

__constant__ int c_din[5]  = {3, 3, 2, 3, 2};   /* ea, em, db, dc, dm */
__constant__ int c_dout[5] = {2, 1, 3, 3, 1};

struct WPtrs {
    const float* w1[5]; const float* b1[5];
    const float* w2[5]; const float* b2[5];
    const float* w3[5]; const float* b3[5];
};

/* ---- prologue: pack f32 weights -> f16, transposed + padded, one contiguous
       46144B block per MLP so a single 1-D TDM descriptor can fetch it ---- */
__global__ void pack_weights(WPtrs P, char* __restrict__ ws) {
    const int m = blockIdx.x;
    const int din = c_din[m], dout = c_dout[m];
    _Float16* w1t = (_Float16*)(ws + (size_t)m * WB_BYTES);   /* [n=128][k=32]  */
    _Float16* w2t = w1t + 4096;                               /* [n=128][k=128] */
    _Float16* w3t = w2t + 16384;                              /* [n=16][k=128]  */
    float*    b   = (float*)(ws + (size_t)m * WB_BYTES + WB_BIAS_OFF);
    for (int i = threadIdx.x; i < 4096; i += blockDim.x) {
        int n = i >> 5, k = i & 31;
        w1t[i] = (_Float16)((k < din) ? P.w1[m][k * HID + n] : 0.0f);
    }
    for (int i = threadIdx.x; i < 16384; i += blockDim.x) {
        int n = i >> 7, k = i & 127;
        w2t[i] = (_Float16)(P.w2[m][k * HID + n]);
    }
    for (int i = threadIdx.x; i < 2048; i += blockDim.x) {
        int n = i >> 7, k = i & 127;
        w3t[i] = (_Float16)((n < dout) ? P.w3[m][k * dout + n] : 0.0f);
    }
    for (int i = threadIdx.x; i < 128; i += blockDim.x) {
        b[i]       = P.b1[m][i];
        b[128 + i] = P.b2[m][i];
    }
    for (int i = threadIdx.x; i < 16; i += blockDim.x)
        b[256 + i] = (i < dout) ? P.b3[m][i] : 0.0f;
}

__device__ __forceinline__ float softplusf(float v) {
    return (v > 0.0f) ? (v + log1pf(expf(-v))) : log1pf(expf(v));
}

/* Tensor Data Mover: async DMA of one packed weight block (global -> LDS).
   1-D tile: data_size=8B, tensor_dim0 = tile_dim0 = WB_QWORDS. Tracked by TENSORcnt. */
__device__ __forceinline__ void tdm_load_weights(const void* gaddr, unsigned lds_off) {
    unsigned long long ga = (unsigned long long)gaddr;
    v4u g0;
    g0[0] = 1u;                                           /* count=1, user mode      */
    g0[1] = lds_off;                                      /* lds_addr (bytes)        */
    g0[2] = (unsigned)(ga & 0xffffffffu);                 /* global_addr[31:0]       */
    g0[3] = (unsigned)((ga >> 32) & 0x01ffffffu)          /* global_addr[56:32]      */
          | 0x80000000u;                                  /* type=2 ("image")        */
    v8u g1;
    g1[0] = 0x00030000u;                                  /* data_size=3 (8B), mask=0 */
    g1[1] = (WB_QWORDS & 0xffffu) << 16;                  /* tensor_dim0[15:0]       */
    g1[2] = (1u << 16) | ((WB_QWORDS >> 16) & 0xffffu);   /* dim1=1 | dim0[31:16]    */
    g1[3] = (WB_QWORDS & 0xffffu) << 16;                  /* tile_dim0               */
    g1[4] = 0u;                                           /* tile_dim1/2 unused      */
    g1[5] = WB_QWORDS;                                    /* tensor_dim0_stride      */
    g1[6] = 0u;
    g1[7] = 0u;
    asm volatile("tensor_load_to_lds %0, %1" :: "s"(g0), "s"(g1) : "memory");
}

/* One full 3-layer MLP for 128 points (16 per wave), all GEMMs via WMMA f16->f32.
   On entry: waits for this phase's TDM weight load, then kicks off the next one. */
__device__ __attribute__((noinline)) void run_mlp(
    unsigned buf_off, const char* next_wts_g, unsigned next_lds_off,
    int tid, int wave, int lane, int din, int incol, int dout, int outcol)
{
    const int row = lane & 15;   /* M row (A) or N column (B/D) within 16-tile */
    const int hb  = lane >> 4;   /* lane half */

    __builtin_amdgcn_s_wait_tensorcnt(0);   /* wave0: this phase's weights landed */
    __syncthreads();                        /* + prior phase done with scal/bufs  */
    if (wave == 0 && next_wts_g)            /* prefetch next phase into other buf */
        tdm_load_weights(next_wts_g, next_lds_off);

    const _Float16* s_w1t = (const _Float16*)(smem + buf_off);
    const _Float16* s_w2t = s_w1t + 4096;
    const _Float16* s_w3t = s_w2t + 16384;
    const float* s_b1 = (const float*)(smem + buf_off + WB_BIAS_OFF);
    const float* s_b2 = s_b1 + 128;
    const float* s_b3 = s_b2 + 128;
    _Float16* h1 = (_Float16*)(smem + SM_H1_OFF) + wave * 2048;  /* [16][128] f16 */
    _Float16* h2 = (_Float16*)(smem + SM_H2_OFF) + wave * 2048;
    float (*s_scal)[SCAL_COLS] = (float(*)[SCAL_COLS])(smem + SM_SCAL_OFF);

    /* ---------- layer 1: [16 x 32pad] @ [32pad x 128] ---------- */
    v16h a1;
    {
        const int prow = wave * 16 + row;
        #pragma unroll
        for (int i = 0; i < 16; ++i) {
            int k = hb * 16 + i;
            float v = (k < din) ? s_scal[prow][incol + k] : 0.0f;
            a1[i] = (_Float16)v;
        }
    }
    #pragma unroll
    for (int t = 0; t < 8; ++t) {
        const v16h b = *(const v16h*)(s_w1t + (t * 16 + row) * 32 + hb * 16);
        float bias = s_b1[t * 16 + row];
        v8f c;
        #pragma unroll
        for (int v = 0; v < 8; ++v) c[v] = bias;
        c = __builtin_amdgcn_wmma_f32_16x16x32_f16(false, a1, false, b, (short)0, c, false, false);
        #pragma unroll
        for (int v = 0; v < 8; ++v)
            h1[(v + 8 * hb) * HID + t * 16 + row] = (_Float16)fmaxf(c[v], 0.0f);
    }
    __syncthreads();

    /* ---------- layer 2: [16 x 128] @ [128 x 128] ---------- */
    #pragma unroll
    for (int t = 0; t < 8; ++t) {
        float bias = s_b2[t * 16 + row];
        v8f c;
        #pragma unroll
        for (int v = 0; v < 8; ++v) c[v] = bias;
        #pragma unroll
        for (int kc = 0; kc < 4; ++kc) {
            const _Float16* ap = h1 + row * HID + kc * 32 + hb * 16;
            v8h alo = *(const v8h*)ap;
            v8h ahi = *(const v8h*)(ap + 8);
            v16h a = __builtin_shufflevector(alo, ahi, 0,1,2,3,4,5,6,7,8,9,10,11,12,13,14,15);
            const v16h b = *(const v16h*)(s_w2t + (t * 16 + row) * HID + kc * 32 + hb * 16);
            c = __builtin_amdgcn_wmma_f32_16x16x32_f16(false, a, false, b, (short)0, c, false, false);
        }
        #pragma unroll
        for (int v = 0; v < 8; ++v)
            h2[(v + 8 * hb) * HID + t * 16 + row] = (_Float16)fmaxf(c[v], 0.0f);
    }
    __syncthreads();

    /* ---------- layer 3: [16 x 128] @ [128 x 16pad] ---------- */
    {
        float bias = s_b3[row];
        v8f c;
        #pragma unroll
        for (int v = 0; v < 8; ++v) c[v] = bias;
        #pragma unroll
        for (int kc = 0; kc < 4; ++kc) {
            const _Float16* ap = h2 + row * HID + kc * 32 + hb * 16;
            v8h alo = *(const v8h*)ap;
            v8h ahi = *(const v8h*)(ap + 8);
            v16h a = __builtin_shufflevector(alo, ahi, 0,1,2,3,4,5,6,7,8,9,10,11,12,13,14,15);
            const v16h b = *(const v16h*)(s_w3t + row * HID + kc * 32 + hb * 16);
            c = __builtin_amdgcn_wmma_f32_16x16x32_f16(false, a, false, b, (short)0, c, false, false);
        }
        if (row < dout) {
            #pragma unroll
            for (int v = 0; v < 8; ++v)
                s_scal[wave * 16 + v + 8 * hb][outcol + row] = c[v];
        }
    }
    __syncthreads();
}

__global__ void __launch_bounds__(256)
autoenc_kernel(const float* __restrict__ x, const float* __restrict__ center,
               const char* __restrict__ wpack, float* __restrict__ out, int npts)
{
    float (*s_scal)[SCAL_COLS] = (float(*)[SCAL_COLS])(smem + SM_SCAL_OFF);

    const int tid  = threadIdx.x;
    const int lane = tid & 31;
    const int wave = tid >> 5;
    const int base = blockIdx.x * 128;
    const float c0 = center[0], c1 = center[1], c2 = center[2];

    /* kick off DMA of phase-0 weights immediately */
    if (wave == 0) tdm_load_weights(wpack, SM_BUF0_OFF);

    /* stage 0: u = unit(x - center), r_p = r^2 (P = 2) */
    if (tid < 128) {
        int p = base + tid;
        float y0 = 0.f, y1 = 0.f, y2 = 0.f;
        if (p < npts) {
            y0 = x[p * 3 + 0] - c0;
            y1 = x[p * 3 + 1] - c1;
            y2 = x[p * 3 + 2] - c2;
        }
        float r2 = y0 * y0 + y1 * y1 + y2 * y2;
        float r  = sqrtf(r2);
        float iv = 1.0f / (r + 1e-8f);
        s_scal[tid][0] = y0 * iv;
        s_scal[tid][1] = y1 * iv;
        s_scal[tid][2] = y2 * iv;
        s_scal[tid][3] = r2;
    }

    /* encoder MLPs on u: ea (dout 2 -> cols 4,5), em (dout 1 -> col 6) */
    run_mlp(SM_BUF0_OFF, wpack + 1 * WB_BYTES, SM_BUF1_OFF,
            tid, wave, lane, 3, 0, 2, 4);
    run_mlp(SM_BUF1_OFF, wpack + 2 * WB_BYTES, SM_BUF0_OFF,
            tid, wave, lane, 3, 0, 1, 6);

    /* pointwise: e=unit(ea), a=softplus(em), w=r_p*a*e, theta, log r_w, r_rec */
    if (tid < 128) {
        float e0 = s_scal[tid][4], e1 = s_scal[tid][5];
        float en = sqrtf(e0 * e0 + e1 * e1) + 1e-8f;
        e0 /= en; e1 /= en;
        float aa = softplusf(s_scal[tid][6]);
        float rp = s_scal[tid][3];
        float w0 = rp * aa * e0, w1 = rp * aa * e1;
        float rw = sqrtf(w0 * w0 + w1 * w1);
        float iw = 1.0f / (rw + 1e-8f);
        s_scal[tid][7]  = w0 * iw;            /* theta0 */
        s_scal[tid][8]  = w1 * iw;            /* theta1 */
        s_scal[tid][9]  = logf(rw + 1e-8f);   /* ci[2]  */
        s_scal[tid][10] = sqrtf(rw);          /* r_rec  */
    }

    /* decoder MLPs: db(theta)->11..13, dc(ci)->14..16, dm(theta)->6 */
    run_mlp(SM_BUF0_OFF, wpack + 3 * WB_BYTES, SM_BUF1_OFF,
            tid, wave, lane, 2, 7, 3, 11);
    run_mlp(SM_BUF1_OFF, wpack + 4 * WB_BYTES, SM_BUF0_OFF,
            tid, wave, lane, 3, 7, 3, 14);
    run_mlp(SM_BUF0_OFF, nullptr, 0,
            tid, wave, lane, 2, 7, 1, 6);

    /* pointwise finale: c = unit(unit(db) + dc), b = softplus(dm), x_hat */
    if (tid < 128) {
        float g0 = s_scal[tid][11], g1 = s_scal[tid][12], g2 = s_scal[tid][13];
        float nb = sqrtf(g0 * g0 + g1 * g1 + g2 * g2) + 1e-8f;
        float s0 = g0 / nb + s_scal[tid][14];
        float s1 = g1 / nb + s_scal[tid][15];
        float s2 = g2 / nb + s_scal[tid][16];
        float ns = sqrtf(s0 * s0 + s1 * s1 + s2 * s2) + 1e-8f;
        float bb = softplusf(s_scal[tid][6]);
        float sc = s_scal[tid][10] * bb / ns;
        int p = base + tid;
        if (p < npts) {
            out[p * 3 + 0] = c0 + sc * s0;
            out[p * 3 + 1] = c1 + sc * s1;
            out[p * 3 + 2] = c2 + sc * s2;
        }
    }
}

extern "C" void kernel_launch(void* const* d_in, const int* in_sizes, int n_in,
                              void* d_out, int out_size, void* d_ws, size_t ws_size,
                              hipStream_t stream) {
    const float* x      = (const float*)d_in[0];
    const float* center = (const float*)d_in[1];
    WPtrs P;
    for (int m = 0; m < 5; ++m) {
        P.w1[m] = (const float*)d_in[2 + 6 * m + 0];
        P.b1[m] = (const float*)d_in[2 + 6 * m + 1];
        P.w2[m] = (const float*)d_in[2 + 6 * m + 2];
        P.b2[m] = (const float*)d_in[2 + 6 * m + 3];
        P.w3[m] = (const float*)d_in[2 + 6 * m + 4];
        P.b3[m] = (const float*)d_in[2 + 6 * m + 5];
    }
    char* ws = (char*)d_ws;

    (void)hipFuncSetAttribute((const void*)autoenc_kernel,
                              hipFuncAttributeMaxDynamicSharedMemorySize, SMEM_BYTES);

    pack_weights<<<5, 256, 0, stream>>>(P, ws);

    const int npts = in_sizes[0] / 3;
    const int nblocks = (npts + 127) / 128;
    autoenc_kernel<<<nblocks, 256, SMEM_BYTES, stream>>>(x, center, ws, (float*)d_out, npts);
}
